// FraudDetector_79482664780219
// MI455X (gfx1250) — compile-verified, matrix operand order checked by reading, not run on hardware
//
#include <hip/hip_runtime.h>
#include <cstdint>
#include <cstddef>
#include <math.h>

#define HID 128
#define NHEAD 2
#define LRELU_SLOPE 0.2f

typedef __attribute__((ext_vector_type(16))) __bf16 v16bf;
typedef __attribute__((ext_vector_type(8)))  __bf16 bf16x8;
typedef __attribute__((ext_vector_type(8)))  float  v8f;

__device__ __forceinline__ __bf16 f2bf(float f) {
  unsigned u = __builtin_bit_cast(unsigned, f);
  unsigned r = (u + 0x7FFFu + ((u >> 16) & 1u)) >> 16;   // round-to-nearest-even
  return __builtin_bit_cast(__bf16, (unsigned short)r);
}

// f32 -> bf16 bulk convert
__global__ void to_bf16(const float* __restrict__ x, __bf16* __restrict__ y, int n) {
  int i = blockIdx.x * blockDim.x + threadIdx.x;
  if (i < n) y[i] = f2bf(x[i]);
}

// Pack B[K,N] f32 into WMMA-fragment order:
// Bp[ (n>>4)*(K>>5)*512 + (k>>5)*512 + (n&15)*32 + (k&31) ]
__global__ void pack_b(const float* __restrict__ B, __bf16* __restrict__ Bp, int K, int N) {
  int i = blockIdx.x * blockDim.x + threadIdx.x;
  if (i >= K * N) return;
  int k = i / N, n = i % N;                  // read coalesced along N
  Bp[(size_t)((n >> 4) * (K >> 5) + (k >> 5)) * 512 + (n & 15) * 32 + (k & 31)] = f2bf(B[i]);
}

// ---------------------------------------------------------------------------
// C[M,N] = A[M,K] @ B[K,N] (+bias)(+relu)(+bf16 copy). A bf16 row-major,
// B pre-packed bf16 fragments. Each wave: one 16-row tile x two 16-col tiles.
// Fragment = two 16B contiguous chunks -> global_load_b128 pairs, no LDS.
// Requires: M % 16 == 0, K % 32 == 0, N % 32 == 0  (both wave tiles always valid).
// ---------------------------------------------------------------------------
__global__ void gemm_wmma_pk(const __bf16* __restrict__ A, const __bf16* __restrict__ Bp,
                             const float* __restrict__ bias, float* __restrict__ C,
                             __bf16* __restrict__ Cbf, int M, int K, int N, int flags) {
  const int lane = threadIdx.x & 31;
  const int wv   = threadIdx.x >> 5;
  const int tm   = blockIdx.x;
  const int hl   = lane >> 4;                 // half-wave
  const int idx  = lane & 15;
  const int tn0  = blockIdx.y * 128 + (2 * wv) * 16;
  const int tn1  = tn0 + 16;
  if (tn0 >= N) return;                       // wave-uniform; no barriers in kernel

  const int row = tm * 16 + idx;
  const __bf16* Arow = A + (size_t)row * K;
  const int kb = K >> 5;
  const __bf16* B0 = Bp + (size_t)(tn0 >> 4) * kb * 512 + idx * 32;
  const __bf16* B1 = Bp + (size_t)(tn1 >> 4) * kb * 512 + idx * 32;

  v8f acc0, acc1;
#pragma unroll
  for (int i = 0; i < 8; ++i) { acc0[i] = 0.f; acc1[i] = 0.f; }

  for (int k0 = 0; k0 < K; k0 += 32) {
    __builtin_prefetch(Arow + k0 + 256, 0, 3);
    bf16x8 alo = *(const bf16x8*)(Arow + k0 + hl * 8);        // K = hl*8 + 0..7
    bf16x8 ahi = *(const bf16x8*)(Arow + k0 + 16 + hl * 8);   // K = 16 + hl*8 + 0..7
    v16bf af = __builtin_shufflevector(alo, ahi, 0, 1, 2, 3, 4, 5, 6, 7,
                                       8, 9, 10, 11, 12, 13, 14, 15);
    {
      const __bf16* bp = B0 + (size_t)(k0 >> 5) * 512;
      bf16x8 blo = *(const bf16x8*)(bp + hl * 8);
      bf16x8 bhi = *(const bf16x8*)(bp + 16 + hl * 8);
      v16bf bfr = __builtin_shufflevector(blo, bhi, 0, 1, 2, 3, 4, 5, 6, 7,
                                          8, 9, 10, 11, 12, 13, 14, 15);
      acc0 = __builtin_amdgcn_wmma_f32_16x16x32_bf16(false, af, false, bfr,
                                                     (short)0, acc0, false, false);
    }
    {
      const __bf16* bp = B1 + (size_t)(k0 >> 5) * 512;
      bf16x8 blo = *(const bf16x8*)(bp + hl * 8);
      bf16x8 bhi = *(const bf16x8*)(bp + 16 + hl * 8);
      v16bf bfr = __builtin_shufflevector(blo, bhi, 0, 1, 2, 3, 4, 5, 6, 7,
                                          8, 9, 10, 11, 12, 13, 14, 15);
      acc1 = __builtin_amdgcn_wmma_f32_16x16x32_bf16(false, af, false, bfr,
                                                     (short)0, acc1, false, false);
    }
  }

  // store D: lane layout VGPR r -> M = r + 8*half, N = idx
#pragma unroll
  for (int r = 0; r < 8; ++r) {
    int m = tm * 16 + hl * 8 + r;
    {
      float v = acc0[r];
      if (flags & 1) v += bias[tn0 + idx];
      if (flags & 2) v = fmaxf(v, 0.f);
      C[(size_t)m * N + tn0 + idx] = v;
      if (flags & 4) Cbf[(size_t)m * N + tn0 + idx] = f2bf(v);
    }
    {
      float v = acc1[r];
      if (flags & 1) v += bias[tn1 + idx];
      if (flags & 2) v = fmaxf(v, 0.f);
      C[(size_t)m * N + tn1 + idx] = v;
      if (flags & 4) Cbf[(size_t)m * N + tn1 + idx] = f2bf(v);
    }
  }
}

// av[h*128+k] = sum_c W[k, h*128+c] * a[h*128+c]   (fold attention vec thru W)
__global__ void fold_attn(const float* __restrict__ W, const float* __restrict__ a,
                          float* __restrict__ av) {
  int t = threadIdx.x;            // 256 threads: h = t>>7, k = t&127
  int h = t >> 7, k = t & 127;
  float s = 0.f;
#pragma unroll 8
  for (int c = 0; c < HID; ++c) s += W[(size_t)k * (NHEAD * HID) + h * HID + c] * a[h * HID + c];
  av[h * HID + k] = s;
}

// al[n,h] = dot(h[n,:], av[h,:])   one wave per node
__global__ void gemv_al(const float* __restrict__ hfeat, const float* __restrict__ av,
                        float* __restrict__ al, int Nn) {
  int n = blockIdx.x * 8 + (threadIdx.x >> 5);
  int lane = threadIdx.x & 31;
  if (n >= Nn) return;
  const float* hp = hfeat + (size_t)n * HID;
  float s0 = 0.f, s1 = 0.f;
#pragma unroll
  for (int c = lane; c < HID; c += 32) {
    float v = hp[c];
    s0 += v * av[c];
    s1 += v * av[HID + c];
  }
#pragma unroll
  for (int off = 16; off; off >>= 1) {
    s0 += __shfl_xor(s0, off, 32);
    s1 += __shfl_xor(s1, off, 32);
  }
  if (lane == 0) { al[(size_t)n * 2] = s0; al[(size_t)n * 2 + 1] = s1; }
}

__global__ void init_f32(float* __restrict__ p, float v, int n) {
  int i = blockIdx.x * blockDim.x + threadIdx.x;
  if (i < n) p[i] = v;
}

__device__ __forceinline__ void atomicMaxF(float* addr, float v) {
  int* ia = (int*)addr;
  int old = __float_as_int(*((volatile float*)addr));
  while (__int_as_float(old) < v) {
    int assumed = old;
    old = atomicCAS(ia, assumed, __float_as_int(v));
    if (old == assumed) break;
  }
}

__device__ __forceinline__ float lrelu(float v) { return v > 0.f ? v : LRELU_SLOPE * v; }

__global__ void edge_max(const int* __restrict__ src, const int* __restrict__ dst,
                         const float* __restrict__ als, const float* __restrict__ ald,
                         float* __restrict__ m, int E) {
  int e = blockIdx.x * blockDim.x + threadIdx.x;
  if (e >= E) return;
  int s = src[e], d = dst[e];
#pragma unroll
  for (int h = 0; h < NHEAD; ++h) {
    float v = lrelu(als[(size_t)s * 2 + h] + ald[(size_t)d * 2 + h]);
    atomicMaxF(&m[(size_t)d * 2 + h], v);
  }
}

__global__ void edge_exp(const int* __restrict__ src, const int* __restrict__ dst,
                         const float* __restrict__ als, const float* __restrict__ ald,
                         const float* __restrict__ m, float* __restrict__ denom,
                         float* __restrict__ ex, int E) {
  int e = blockIdx.x * blockDim.x + threadIdx.x;
  if (e >= E) return;
  int s = src[e], d = dst[e];
#pragma unroll
  for (int h = 0; h < NHEAD; ++h) {
    float v = lrelu(als[(size_t)s * 2 + h] + ald[(size_t)d * 2 + h]);
    float x = expf(v - m[(size_t)d * 2 + h]);
    ex[(size_t)e * 2 + h] = x;
    atomicAdd(&denom[(size_t)d * 2 + h], x);
  }
}

// acc[dst,c] += 0.5*(alpha0*xs[src,c] + alpha1*xs[src,128+c])  (head mean folded)
__global__ void edge_aggr(const int* __restrict__ src, const int* __restrict__ dst,
                          const float* __restrict__ ex, const float* __restrict__ denom,
                          const float* __restrict__ xs, float* __restrict__ acc, int E) {
  int e = blockIdx.x * 8 + (threadIdx.x >> 5);
  int lane = threadIdx.x & 31;
  if (e >= E) return;
  int s = src[e], d = dst[e];
  float a0 = ex[(size_t)e * 2 + 0] / (denom[(size_t)d * 2 + 0] + 1e-16f);
  float a1 = ex[(size_t)e * 2 + 1] / (denom[(size_t)d * 2 + 1] + 1e-16f);
  const float* x = xs + (size_t)s * (NHEAD * HID);
  float* o = acc + (size_t)d * HID;
#pragma unroll
  for (int c = lane; c < HID; c += 32)
    atomicAdd(&o[c], 0.5f * (a0 * x[c] + a1 * x[HID + c]));
}

__global__ void bias_add(float* __restrict__ acc, const float* __restrict__ b, int Nn) {
  size_t i = (size_t)blockIdx.x * blockDim.x + threadIdx.x;
  if (i >= (size_t)Nn * HID) return;
  acc[i] += b[i & (HID - 1)];
}

// h = relu(a); hbf = bf16(h)
__global__ void relu_copy_bf(const float* __restrict__ a, float* __restrict__ h,
                             __bf16* __restrict__ hbf, int n) {
  int i = blockIdx.x * blockDim.x + threadIdx.x;
  if (i < n) {
    float v = fmaxf(a[i], 0.f);
    h[i] = v;
    hbf[i] = f2bf(v);
  }
}

// out[n] = dot(Z[n,0:64], w) + b   one wave per row
__global__ void rowdot64(const float* __restrict__ Z, const float* __restrict__ w,
                         const float* __restrict__ b, float* __restrict__ out, int M) {
  int n = blockIdx.x * 8 + (threadIdx.x >> 5);
  int lane = threadIdx.x & 31;
  if (n >= M) return;
  const float* z = Z + (size_t)n * 64;
  float s = z[lane] * w[lane] + z[32 + lane] * w[32 + lane];
#pragma unroll
  for (int off = 16; off; off >>= 1) s += __shfl_xor(s, off, 32);
  if (lane == 0) out[n] = s + b[0];
}

// ---------------------------------------------------------------------------
extern "C" void kernel_launch(void* const* d_in, const int* in_sizes, int n_in,
                              void* d_out, int out_size, void* d_ws, size_t ws_size,
                              hipStream_t stream) {
  const float* post_cls = (const float*)d_in[0];
  const float* user_x   = (const float*)d_in[1];
  const float* ent_x    = (const float*)d_in[2];
  const float* Wpost = (const float*)d_in[3];
  const float* bpost = (const float*)d_in[4];
  const float* Wuser = (const float*)d_in[5];
  const float* buser = (const float*)d_in[6];
  const float* Went  = (const float*)d_in[7];
  const float* bent  = (const float*)d_in[8];
  const float* gWsrc = (const float*)d_in[9];   // [2,5,128,256]
  const float* gWdst = (const float*)d_in[10];  // [2,5,128,256]
  const float* gasrc = (const float*)d_in[11];  // [2,5,2,128]
  const float* gadst = (const float*)d_in[12];  // [2,5,2,128]
  const float* gbias = (const float*)d_in[13];  // [2,5,128]
  const float* Wc1 = (const float*)d_in[14];
  const float* bc1 = (const float*)d_in[15];
  const float* Wc2 = (const float*)d_in[16];
  const float* bc2 = (const float*)d_in[17];

  const int NU = 100000, NP = 50000, NEN = 20000;
  const int TEXT_D = 768, USER_D = 64, ENT_D = 64;

  // workspace carve-up
  size_t off = 0;
  auto carve = [&](size_t bytes) -> void* {
    void* p = (char*)d_ws + off;
    off += (bytes + 255) & ~(size_t)255;
    return p;
  };
  float*  h_user   = (float*)carve((size_t)NU * HID * 4);
  float*  h_post   = (float*)carve((size_t)NP * HID * 4);
  float*  h_ent    = (float*)carve((size_t)NEN * HID * 4);
  float*  acc_user = (float*)carve((size_t)NU * HID * 4);
  float*  acc_post = (float*)carve((size_t)NP * HID * 4);
  float*  acc_ent  = (float*)carve((size_t)NEN * HID * 4);
  float*  xs       = (float*)carve((size_t)NU * NHEAD * HID * 4);
  float*  al_s     = (float*)carve((size_t)NU * 2 * 4);
  float*  al_d     = (float*)carve((size_t)NU * 2 * 4);
  float*  mbuf     = (float*)carve((size_t)NU * 2 * 4);
  float*  dbuf     = (float*)carve((size_t)NU * 2 * 4);
  float*  exbuf    = (float*)carve((size_t)500000 * 2 * 4);
  float*  av       = (float*)carve(512 * 4);
  float*  zbuf     = (float*)carve((size_t)NP * 64 * 4);
  __bf16* pc_bf    = (__bf16*)carve((size_t)NP * TEXT_D * 2);
  __bf16* ux_bf    = (__bf16*)carve((size_t)NU * USER_D * 2);
  __bf16* en_bf    = (__bf16*)carve((size_t)NEN * ENT_D * 2);
  __bf16* hu_bf    = (__bf16*)carve((size_t)NU * HID * 2);
  __bf16* hp_bf    = (__bf16*)carve((size_t)NP * HID * 2);
  __bf16* he_bf    = (__bf16*)carve((size_t)NEN * HID * 2);
  __bf16* bpk      = (__bf16*)carve((size_t)TEXT_D * (NHEAD * HID) * 2);  // packed B scratch
  if (off > ws_size) return;                    // workspace too small: bail safely

  auto gemm = [&](const __bf16* A, const float* B, const float* bias, float* C,
                  __bf16* Cbf, int M, int K, int N, int flags) {
    pack_b<<<(K * N + 255) / 256, 256, 0, stream>>>(B, bpk, K, N);
    dim3 grid((M + 15) / 16, (N + 127) / 128);
    gemm_wmma_pk<<<grid, 128, 0, stream>>>(A, bpk, bias, C, Cbf, M, K, N, flags);
  };

  // ---- convert raw inputs to bf16 once -----------------------------------
  to_bf16<<<((size_t)NP * TEXT_D + 255) / 256, 256, 0, stream>>>(post_cls, pc_bf, NP * TEXT_D);
  to_bf16<<<((size_t)NU * USER_D + 255) / 256, 256, 0, stream>>>(user_x, ux_bf, NU * USER_D);
  to_bf16<<<((size_t)NEN * ENT_D + 255) / 256, 256, 0, stream>>>(ent_x, en_bf, NEN * ENT_D);

  // ---- input projections (emit f32 + bf16 copies of h) -------------------
  gemm(pc_bf, Wpost, bpost, h_post, hp_bf, NP, TEXT_D, HID, 1 | 4);
  gemm(ux_bf, Wuser, buser, h_user, hu_bf, NU, USER_D, HID, 1 | 4);
  gemm(en_bf, Went,  bent,  h_ent,  he_bf, NEN, ENT_D, HID, 1 | 4);

  struct Rel { float* hs; __bf16* hsbf; float* hd; float* acc;
               int Ns, Nd, E; const int* s; const int* d; };
  Rel rel[5] = {
    { h_user, hu_bf, h_post, acc_post, NU, NP,  150000, (const int*)d_in[18], (const int*)d_in[19] },
    { h_user, hu_bf, h_post, acc_post, NU, NP,  300000, (const int*)d_in[20], (const int*)d_in[21] },
    { h_post, hp_bf, h_ent,  acc_ent,  NP, NEN, 200000, (const int*)d_in[22], (const int*)d_in[23] },
    { h_user, hu_bf, h_user, acc_user, NU, NU,  500000, (const int*)d_in[24], (const int*)d_in[25] },
    { h_user, hu_bf, h_user, acc_user, NU, NU,  500000, (const int*)d_in[26], (const int*)d_in[27] },
  };

  for (int l = 0; l < 2; ++l) {
    init_f32<<<((size_t)NP * HID + 255) / 256, 256, 0, stream>>>(acc_post, 0.f, NP * HID);
    init_f32<<<((size_t)NU * HID + 255) / 256, 256, 0, stream>>>(acc_user, 0.f, NU * HID);
    init_f32<<<((size_t)NEN * HID + 255) / 256, 256, 0, stream>>>(acc_ent, 0.f, NEN * HID);

    for (int r = 0; r < 5; ++r) {
      Rel& R = rel[r];
      size_t wofs = (size_t)(l * 5 + r) * HID * (NHEAD * HID);
      size_t aofs = (size_t)(l * 5 + r) * NHEAD * HID;
      const float* Ws = gWsrc + wofs;
      const float* Wd = gWdst + wofs;
      const float* as_ = gasrc + aofs;
      const float* ad_ = gadst + aofs;
      const float* bs = gbias + (size_t)(l * 5 + r) * HID;

      fold_attn<<<1, 256, 0, stream>>>(Ws, as_, av);
      fold_attn<<<1, 256, 0, stream>>>(Wd, ad_, av + NHEAD * HID);
      gemm(R.hsbf, Ws, nullptr, xs, nullptr, R.Ns, HID, NHEAD * HID, 0);
      gemv_al<<<(R.Ns + 7) / 8, 256, 0, stream>>>(R.hs, av, al_s, R.Ns);
      gemv_al<<<(R.Nd + 7) / 8, 256, 0, stream>>>(R.hd, av + NHEAD * HID, al_d, R.Nd);
      init_f32<<<(R.Nd * 2 + 255) / 256, 256, 0, stream>>>(mbuf, -INFINITY, R.Nd * 2);
      init_f32<<<(R.Nd * 2 + 255) / 256, 256, 0, stream>>>(dbuf, 0.f, R.Nd * 2);
      edge_max<<<(R.E + 255) / 256, 256, 0, stream>>>(R.s, R.d, al_s, al_d, mbuf, R.E);
      edge_exp<<<(R.E + 255) / 256, 256, 0, stream>>>(R.s, R.d, al_s, al_d, mbuf, dbuf, exbuf, R.E);
      edge_aggr<<<(R.E + 7) / 8, 256, 0, stream>>>(R.s, R.d, exbuf, dbuf, xs, R.acc, R.E);
      bias_add<<<(((size_t)R.Nd * HID) + 255) / 256, 256, 0, stream>>>(R.acc, bs, R.Nd);
    }

    relu_copy_bf<<<((size_t)NP * HID + 255) / 256, 256, 0, stream>>>(acc_post, h_post, hp_bf, NP * HID);
    relu_copy_bf<<<((size_t)NU * HID + 255) / 256, 256, 0, stream>>>(acc_user, h_user, hu_bf, NU * HID);
    relu_copy_bf<<<((size_t)NEN * HID + 255) / 256, 256, 0, stream>>>(acc_ent, h_ent, he_bf, NEN * HID);
  }

  // ---- classifier --------------------------------------------------------
  gemm(hp_bf, Wc1, bc1, zbuf, nullptr, NP, HID, 64, 3);      // relu(h@Wc1+bc1)
  rowdot64<<<(NP + 7) / 8, 256, 0, stream>>>(zbuf, Wc2, bc2, (float*)d_out, NP);
}
